// BrainDecayAttention_8366596293036
// MI455X (gfx1250) — matrix operand order repaired and checked by reading, not compiled, not run on hardware
//
#include <hip/hip_runtime.h>

#define Bc 16
#define Hc 8
#define Nc 1024
#define Dc 64
#define HHc 4

typedef __attribute__((ext_vector_type(16))) _Float16 v16h;
typedef __attribute__((ext_vector_type(8)))  float    v8f;
typedef __attribute__((ext_vector_type(4)))  float    f4;
typedef int v4i_ __attribute__((vector_size(16)));   // matches async builtin's pointee

#define WAVES   4
#define TILE_M  64          // 4 waves x 16 query rows
#define KV_PAD  68          // f32 row stride of staged K/V tiles (bank spread, 16B aligned)
#define Q_PAD   72          // f16 row stride of staged Q tile
#define S_PAD   1028        // f32 row stride of logits buffer

// LDS layout (float units)
#define OFF_KV     0
#define KBUF_F     (16 * KV_PAD)                 // 1088 floats per K buffer
#define VBUF_F     (32 * KV_PAD)                 // 2176 floats per V buffer
#define OFF_STATS  (OFF_KV + 2 * VBUF_F)         // 4352
#define OFF_Q      (OFF_STATS + WAVES * 32)      // 4480 (region holds f16)
#define OFF_S      (OFF_Q + (WAVES * 16 * Q_PAD) / 2)
#define LDS_FLOATS (OFF_S + WAVES * 16 * S_PAD)
#define LDS_BYTES  (LDS_FLOATS * 4)

#if __has_builtin(__builtin_amdgcn_global_load_async_to_lds_b128) && \
    __has_builtin(__builtin_amdgcn_s_wait_asynccnt)
#define HAS_ASYNC 1
#else
#define HAS_ASYNC 0
#endif

__device__ __forceinline__ void stage16(const float* g, float* l) {
#if HAS_ASYNC
  __builtin_amdgcn_global_load_async_to_lds_b128(
      (__attribute__((address_space(1))) v4i_*)(void*)g,
      (__attribute__((address_space(3))) v4i_*)(void*)l, 0, 0);
#else
  *(f4*)l = *(const f4*)g;
#endif
}

template <int PENDING>
__device__ __forceinline__ void stage_wait() {
#if HAS_ASYNC
  __builtin_amdgcn_s_wait_asynccnt(PENDING);
#endif
}

template <bool IS_SHORT>
__global__ __launch_bounds__(128) void bda_kernel(
    const float* __restrict__ q, const float* __restrict__ k,
    const float* __restrict__ v, const float* __restrict__ sph,
    const float* __restrict__ hop, const float* __restrict__ gam,
    float* __restrict__ out, float* __restrict__ wout, int hoff) {
  extern __shared__ float smem[];
  float*    ldsKV  = smem + OFF_KV;
  float*    stats  = smem + OFF_STATS;
  _Float16* ldsQ   = (_Float16*)(smem + OFF_Q);
  float*    ldsS   = smem + OFF_S;

  const int tid  = threadIdx.x;
  const int lane = tid & 31;
  const int wave = tid >> 5;
  const int n16  = lane & 15;   // WMMA B/C/D column (or A row)
  const int hh   = lane >> 4;   // lane half selector in WMMA layouts

  const int qtiles = Nc / TILE_M;                 // 16
  const int qt   = blockIdx.x % qtiles;
  const int rem  = blockIdx.x / qtiles;
  const int hloc = rem % HHc;                     // 0..3 within this half of the heads
  const int b    = rem / HHc;
  const int h    = hoff + hloc;
  const int bh   = b * Hc + h;
  const int qbase = qt * TILE_M + wave * 16;      // this wave's first query row

  const size_t headOff = (size_t)bh * (Nc * Dc);
  const float* qg   = q + headOff + (size_t)qbase * Dc;
  const float* kg   = k + headOff;
  const float* vg   = v + headOff;
  const float* sphB = sph + (size_t)b * Nc * Nc;

  float hopv = 0.f, lng = 0.f;
  if (IS_SHORT) {
    hopv = hop[hloc];
    float g = 1.f / (1.f + __expf(-gam[hloc]));   // sigmoid(short_gamma)
    lng = __logf(g);                              // ln(gamma)
  }

  // ---- stage this wave's 16x64 Q tile to LDS as f16 ----
  _Float16* myQ = ldsQ + wave * 16 * Q_PAD;
  for (int idx = lane * 4; idx < 16 * Dc; idx += 32 * 4) {
    f4 fv = *(const f4*)(qg + idx);
    int row = idx >> 6, col = idx & 63;
    _Float16* dp = myQ + row * Q_PAD + col;
    dp[0] = (_Float16)fv.x; dp[1] = (_Float16)fv.y;
    dp[2] = (_Float16)fv.z; dp[3] = (_Float16)fv.w;
  }
  __syncthreads();

  // ---- build Q A-operands (16x32 f16 each, kept in VGPRs for whole kernel) ----
  v16h aq0, aq1;
#pragma unroll
  for (int j = 0; j < 16; ++j) {
    int d = 16 * hh + j;
    aq0[j] = myQ[n16 * Q_PAD + d];
    aq1[j] = myQ[n16 * Q_PAD + 32 + d];
  }

  float rmax[8], rsum[8];
#pragma unroll
  for (int r = 0; r < 8; ++r) { rmax[r] = -3.0e30f; rsum[r] = 0.f; }

  float* myS = ldsS + wave * 16 * S_PAD;

  // ---- prologue: async-stage K tile 0 (double buffered) ----
  {
    int row = tid >> 3, col = (tid & 7) * 8;
    const float* src = kg + (size_t)row * Dc + col;
    float* dst = ldsKV + row * KV_PAD + col;
    stage16(src, dst);
    stage16(src + 4, dst + 4);
  }

  // ==== phase 1: S = (Q Kt)*scale (+decay modulation), logits -> LDS, online softmax ====
  const int KT = Nc / 16;                          // 64 key tiles
  for (int kt = 0; kt < KT; ++kt) {
    float* kbuf = ldsKV + (kt & 1) * KBUF_F;

    // batch-issue sph loads for this tile first so they fly under the WMMAs
    float spv[8];
    if (IS_SHORT) {
      const float* sphP = sphB + (size_t)(qbase + 8 * hh) * Nc + kt * 16 + n16;
#pragma unroll
      for (int r = 0; r < 8; ++r) spv[r] = sphP[(size_t)r * Nc];
    }

    if (kt + 1 < KT) {                             // overlap copy of kt+1 with compute of kt
      int row = tid >> 3, col = (tid & 7) * 8;
      const float* src = kg + (size_t)((kt + 1) * 16 + row) * Dc + col;
      float* dst = ldsKV + ((kt + 1) & 1) * KBUF_F + row * KV_PAD + col;
      stage16(src, dst);
      stage16(src + 4, dst + 4);
      stage_wait<2>();                             // tile kt complete; kt+1 in flight
    } else {
      stage_wait<0>();
    }
    __syncthreads();

    v16h bk0, bk1;                                 // Kt as 32x16 B-operands
#pragma unroll
    for (int j = 0; j < 16; ++j) {
      int d = 16 * hh + j;
      bk0[j] = (_Float16)kbuf[n16 * KV_PAD + d];
      bk1[j] = (_Float16)kbuf[n16 * KV_PAD + 32 + d];
    }

    v8f s = {};
    s = __builtin_amdgcn_wmma_f32_16x16x32_f16(false, aq0, false, bk0, (short)0, s, false, false);
    s = __builtin_amdgcn_wmma_f32_16x16x32_f16(false, aq1, false, bk1, (short)0, s, false, false);

#pragma unroll
    for (int r = 0; r < 8; ++r) {
      float logit = s[r] * 0.125f;                 // 1/sqrt(64)
      if (IS_SHORT) {
        float dec = spv[r] - hopv;
        dec = dec > 0.f ? dec : 0.f;
        logit *= __expf(dec * lng);                // gamma^relu(sph-hop)
      }
      myS[(r + 8 * hh) * S_PAD + kt * 16 + n16] = logit;
      float m0 = rmax[r];
      float mn = m0 > logit ? m0 : logit;          // online softmax
      rsum[r] = rsum[r] * __expf(m0 - mn) + __expf(logit - mn);
      rmax[r] = mn;
    }
    __syncthreads();                               // protect buffer reuse
  }

  // ---- reduce softmax stats across the 16 key-column lanes of each half ----
#pragma unroll
  for (int d = 1; d < 16; d <<= 1) {
#pragma unroll
    for (int r = 0; r < 8; ++r) {
      float om = __shfl_xor(rmax[r], d, 32);
      float os = __shfl_xor(rsum[r], d, 32);
      float mn = rmax[r] > om ? rmax[r] : om;
      rsum[r] = rsum[r] * __expf(rmax[r] - mn) + os * __expf(om - mn);
      rmax[r] = mn;
    }
  }
  float* myStats = stats + wave * 32;
  if (n16 == 0) {
#pragma unroll
    for (int r = 0; r < 8; ++r) {
      myStats[r + 8 * hh]      = rmax[r];
      myStats[16 + r + 8 * hh] = 1.0f / rsum[r];
    }
  }
  __syncthreads();

  // ---- prologue: async-stage V chunk 0 (overlaps with W streaming below) ----
  {
    int row = tid >> 2, col = (tid & 3) * 16;
    const float* src = vg + (size_t)row * Dc + col;
    float* dst = ldsKV + row * KV_PAD + col;
    stage16(src, dst);       stage16(src + 4, dst + 4);
    stage16(src + 8, dst + 8); stage16(src + 12, dst + 12);
  }

  // ==== phase 2: stream normalized W to global (nontemporal: 537 MB, don't pollute L2) ====
  float* wRow = wout + ((size_t)bh * Nc + qbase) * Nc;
  for (int r = 0; r < 16; ++r) {
    float mx = myStats[r], inv = myStats[16 + r];
    float* wr = wRow + (size_t)r * Nc;
    for (int n0 = lane * 4; n0 < Nc; n0 += 128) {
      f4 sv = *(const f4*)(myS + r * S_PAD + n0);
      f4 wv;
      wv.x = __expf(sv.x - mx) * inv;
      wv.y = __expf(sv.y - mx) * inv;
      wv.z = __expf(sv.z - mx) * inv;
      wv.w = __expf(sv.w - mx) * inv;
      __builtin_nontemporal_store(wv, (f4*)(wr + n0));
    }
  }

  // ==== phase 3: O = W V via WMMA, V chunks of 32 keys, double buffered ====
  const float myMx  = myStats[n16];
  const float myInv = myStats[16 + n16];
  v8f acc[4] = {};                                 // 4 column tiles of D=64

  const int VT = Nc / 32;                          // 32 value chunks
  for (int nt = 0; nt < VT; ++nt) {
    float* vbuf = ldsKV + (nt & 1) * VBUF_F;
    if (nt + 1 < VT) {
      int row = tid >> 2, col = (tid & 3) * 16;
      const float* src = vg + (size_t)((nt + 1) * 32 + row) * Dc + col;
      float* dst = ldsKV + ((nt + 1) & 1) * VBUF_F + row * KV_PAD + col;
      stage16(src, dst);       stage16(src + 4, dst + 4);
      stage16(src + 8, dst + 8); stage16(src + 12, dst + 12);
      stage_wait<4>();
    } else {
      stage_wait<0>();
    }
    __syncthreads();

    v16h aw;                                       // W chunk as 16x32 f16 A-operand
#pragma unroll
    for (int j = 0; j < 16; ++j) {
      int n = nt * 32 + 16 * hh + j;
      float sv = myS[n16 * S_PAD + n];
      aw[j] = (_Float16)(__expf(sv - myMx) * myInv);
    }
#pragma unroll
    for (int dt = 0; dt < 4; ++dt) {
      v16h bv;                                     // V chunk as 32x16 B-operand
#pragma unroll
      for (int j = 0; j < 16; ++j)
        bv[j] = (_Float16)vbuf[(16 * hh + j) * KV_PAD + dt * 16 + n16];
      acc[dt] = __builtin_amdgcn_wmma_f32_16x16x32_f16(false, aw, false, bv,
                                                       (short)0, acc[dt], false, false);
    }
    __syncthreads();
  }

  // ---- store O ----
  float* oRow = out + ((size_t)bh * Nc + qbase) * Dc;
#pragma unroll
  for (int dt = 0; dt < 4; ++dt)
#pragma unroll
    for (int r = 0; r < 8; ++r)
      __builtin_nontemporal_store(acc[dt][r],
          oRow + (size_t)(r + 8 * hh) * Dc + dt * 16 + n16);
}

extern "C" void kernel_launch(void* const* d_in, const int* in_sizes, int n_in,
                              void* d_out, int out_size, void* d_ws, size_t ws_size,
                              hipStream_t stream) {
  (void)in_sizes; (void)n_in; (void)d_ws; (void)ws_size; (void)out_size;
  const float* q   = (const float*)d_in[0];
  const float* k   = (const float*)d_in[1];
  const float* v   = (const float*)d_in[2];
  const float* sph = (const float*)d_in[3];
  const float* hop = (const float*)d_in[4];
  const float* gam = (const float*)d_in[5];
  float* out  = (float*)d_out;
  float* wout = out + (size_t)Bc * Hc * Nc * Dc;   // outputs concatenated: O then W

  (void)hipFuncSetAttribute(reinterpret_cast<const void*>(bda_kernel<true>),
                            hipFuncAttributeMaxDynamicSharedMemorySize, LDS_BYTES);
  (void)hipFuncSetAttribute(reinterpret_cast<const void*>(bda_kernel<false>),
                            hipFuncAttributeMaxDynamicSharedMemorySize, LDS_BYTES);

  dim3 grid(Bc * HHc * (Nc / TILE_M));             // 1024 blocks per half
  // heads 0..3: decay-modulated attention
  bda_kernel<true><<<grid, 128, LDS_BYTES, stream>>>(q, k, v, sph, hop, gam, out, wout, 0);
  // heads 4..7: plain softmax attention (no sph traffic at all)
  bda_kernel<false><<<grid, 128, LDS_BYTES, stream>>>(q, k, v, sph, hop, gam, out, wout, HHc);
}
